// BasalGanglia_49357764166295
// MI455X (gfx1250) — matrix operand order repaired and checked by reading, not compiled, not run on hardware
//
#include <hip/hip_runtime.h>
#include <hip/hip_bf16.h>

typedef __attribute__((ext_vector_type(16))) _Float16 v16h;
typedef __attribute__((ext_vector_type(8)))  _Float16 v8h;
typedef __attribute__((ext_vector_type(8)))  float    v8f;

static constexpr int BATCH = 8192;
static constexpr int KP    = 320;   // 300 padded to multiple of 32
static constexpr int NSLAT = 304;   // 300 padded to multiple of 16

// ---- workspace byte offsets (all 16B aligned) ----
static constexpr size_t OFF_STIM   = 0;                                   // [B][320] f16
static constexpr size_t OFF_VF0    = OFF_STIM   + (size_t)BATCH*KP*2;     // [64][320] f16
static constexpr size_t OFF_VF1    = OFF_VF0    + (size_t)64*KP*2;        // [128][64] f16
static constexpr size_t OFF_VF2    = OFF_VF1    + (size_t)128*64*2;       // [64][128] f16
static constexpr size_t OFF_JKD    = OFF_VF2    + (size_t)64*128*2;       // [128][320] f16 (jd1|jd2|kd1|kd2)
static constexpr size_t OFF_SG     = OFF_JKD    + (size_t)128*KP*2;       // [32][320] f16
static constexpr size_t OFF_GS     = OFF_SG     + (size_t)32*KP*2;        // [304][32] f16
static constexpr size_t OFF_GLAT   = OFF_GS     + (size_t)NSLAT*32*2;     // [32][32] f16
static constexpr size_t OFF_SLAT   = OFF_GLAT   + (size_t)32*32*2;        // [304][320] f16
static constexpr size_t OFF_STNGPI = OFF_SLAT   + (size_t)NSLAT*KP*2;     // [16][320] f16
static constexpr size_t OFF_LSTM   = OFF_STNGPI + (size_t)16*KP*2;        // [80][32] f16 (ih|hh packed in K)
static constexpr size_t OFF_VD2    = OFF_LSTM   + (size_t)80*32*2;        // [B][20] f32
static constexpr size_t OFF_DP     = OFF_VD2    + (size_t)BATCH*20*4;     // [B][2] f32
static constexpr size_t OFF_LAM    = OFF_DP     + (size_t)BATCH*2*4;      // [B] f32

// ---- stn kernel dynamic-LDS layout (byte offsets from dynamic-LDS base) ----
static constexpr int LDS_SLAT  = 0;        // [304][320] f16 = 194560
static constexpr int LDS_VSTN  = 194560;   // [32][320] f16  = 20480
static constexpr int LDS_XSTN  = 215040;   // [32][304] f32  = 38912
static constexpr int LDS_GATES = 253952;   // [32][80] f32   = 10240
static constexpr int LDS_XGPE  = 264192;   // [32][32] f16   = 2048
static constexpr int LDS_XH    = 266240;   // [32][32] f16   = 2048
static constexpr int LDS_C     = 268288;   // [32][20] f32   = 2560
static constexpr int LDS_H     = 270848;   // [32][20] f32   = 2560
static constexpr int LDS_VD2   = 273408;   // [32][20] f32   = 2560
static constexpr int LDS_VGPI  = 275968;   // [32][2] f32    = 256
static constexpr int LDS_DP    = 276224;   // [32][2] f32    = 256
static constexpr int LDS_LAM   = 276480;   // [32] f32       = 128
static constexpr int STN_LDS   = 276608;

__device__ __forceinline__ float fast_tanh(float x) {
#if __has_builtin(__builtin_amdgcn_tanhf)
  return __builtin_amdgcn_tanhf(x);      // v_tanh_f32 (CDNA5 TRANS op)
#else
  return tanhf(x);
#endif
}
__device__ __forceinline__ float sigmoidf_(float x) { return 1.0f / (1.0f + __expf(-x)); }

// A fragment: 16x32 f16, row-major source with given K-stride (elements)
__device__ __forceinline__ v16h load_a_frag(const _Float16* A, int stride, int m0, int k0) {
  const int lane = threadIdx.x & 31;
  const int row  = m0 + (lane & 15);
  const int kb   = (lane < 16) ? 0 : 8;
  const _Float16* p = A + (size_t)row * stride + k0;
  v8h lo = *(const v8h*)(p + kb);
  v8h hi = *(const v8h*)(p + 16 + kb);
  v16h r;
#pragma unroll
  for (int i = 0; i < 8; ++i) { r[i] = lo[i]; r[8 + i] = hi[i]; }
  return r;
}

// B fragment: 32x16 f16; weights stored as W[n][k] row-major (lane = column n)
__device__ __forceinline__ v16h load_b_frag(const _Float16* W, int stride, int n0, int k0) {
  const int lane = threadIdx.x & 31;
  const int row  = n0 + (lane & 15);
  const int kb   = (lane < 16) ? 0 : 8;
  const _Float16* p = W + (size_t)row * stride + k0;
  v8h lo = *(const v8h*)(p + kb);
  v8h hi = *(const v8h*)(p + 16 + kb);
  v16h r;
#pragma unroll
  for (int i = 0; i < 8; ++i) { r[i] = lo[i]; r[8 + i] = hi[i]; }
  return r;
}

__device__ __forceinline__ v8f wmma16(v16h a, v16h b, v8f c) {
  return __builtin_amdgcn_wmma_f32_16x16x32_f16(false, a, false, b, (short)0, c, false, false);
}

__device__ __forceinline__ void padw(const float* __restrict__ W, _Float16* __restrict__ D,
                                     int o, int i, int oP, int iP, int tid, int np) {
  const int n = oP * iP;
  for (int x = tid; x < n; x += np) {
    int r = x / iP, k = x - r * iP;
    D[x] = (r < o && k < i) ? (_Float16)W[r * i + k] : (_Float16)0.0f;
  }
}

// ---------------------------------------------------------------------------
__global__ void bg_prep(const float* __restrict__ stim,
                        const float* __restrict__ wvf0, const float* __restrict__ wvf1,
                        const float* __restrict__ wvf2,
                        const float* __restrict__ wjd1, const float* __restrict__ wjd2,
                        const float* __restrict__ wkd1, const float* __restrict__ wkd2,
                        const float* __restrict__ wsg,  const float* __restrict__ wgs,
                        const float* __restrict__ wglat,const float* __restrict__ wslat,
                        const float* __restrict__ wstngpi,
                        const float* __restrict__ wih,  const float* __restrict__ whh,
                        unsigned char* __restrict__ ws) {
  const int tid = blockIdx.x * blockDim.x + threadIdx.x;
  const int np  = gridDim.x * blockDim.x;
  { _Float16* d = (_Float16*)(ws + OFF_STIM);
    for (int x = tid; x < BATCH * KP; x += np) {
      int r = x / KP, k = x - r * KP;
      d[x] = (k < 300) ? (_Float16)stim[r * 300 + k] : (_Float16)0.0f;
    } }
  padw(wvf0,   (_Float16*)(ws + OFF_VF0),     64, 300,  64,  KP, tid, np);
  padw(wvf1,   (_Float16*)(ws + OFF_VF1),    128,  64, 128,  64, tid, np);
  padw(wvf2,   (_Float16*)(ws + OFF_VF2),     64, 128,  64, 128, tid, np);
  padw(wsg,    (_Float16*)(ws + OFF_SG),      20, 300,  32,  KP, tid, np);
  padw(wgs,    (_Float16*)(ws + OFF_GS),     300,  20, NSLAT, 32, tid, np);
  padw(wglat,  (_Float16*)(ws + OFF_GLAT),    20,  20,  32,  32, tid, np);
  padw(wslat,  (_Float16*)(ws + OFF_SLAT),   300, 300, NSLAT, KP, tid, np);
  padw(wstngpi,(_Float16*)(ws + OFF_STNGPI),   2, 300,  16,  KP, tid, np);
  { _Float16* d = (_Float16*)(ws + OFF_JKD);
    for (int x = tid; x < 128 * KP; x += np) {
      int r = x / KP, k = x - r * KP;
      int blk = r >> 5, rr = r & 31;
      float v = 0.0f;
      if (rr < 20 && k < 300) {
        const float* W = (blk == 0) ? wjd1 : (blk == 1) ? wjd2 : (blk == 2) ? wkd1 : wkd2;
        v = W[rr * 300 + k];
      }
      d[x] = (_Float16)v;
    } }
  { _Float16* d = (_Float16*)(ws + OFF_LSTM);
    for (int x = tid; x < 80 * 32; x += np) {
      int r = x >> 5, k = x & 31;
      float v = 0.0f;
      if (k < 2) v = wih[r * 2 + k];
      else if (k < 22) v = whh[r * 20 + (k - 2)];
      d[x] = (_Float16)v;
    } }
}

// ---------------------------------------------------------------------------
// Value head: 300 -> 64 -> 128 -> 64 -> 1 (tanh). One wave per 16-row tile.
__global__ void __launch_bounds__(256) bg_vf(const unsigned char* __restrict__ ws,
                     const float* __restrict__ bvf0, const float* __restrict__ bvf1,
                     const float* __restrict__ bvf2,
                     const float* __restrict__ wvf3, const float* __restrict__ bvf3,
                     float* __restrict__ vt_out) {
  extern __shared__ unsigned char smem_vf[];
  const int lane = threadIdx.x & 31;
  const int wid  = threadIdx.x >> 5;
  const int rowbase = blockIdx.x * 128 + wid * 16;
  _Float16* buf0 = (_Float16*)(smem_vf + wid * 6144);          // [16][64]
  _Float16* buf1 = (_Float16*)(smem_vf + wid * 6144 + 2048);   // [16][128]
  const _Float16* stimh = (const _Float16*)(ws + OFF_STIM) + (size_t)rowbase * KP;
  const int c = lane & 15, hi8 = (lane >> 4) * 8;
  __builtin_prefetch(ws + OFF_VF0, 0, 0);
  __builtin_prefetch(ws + OFF_VF1, 0, 0);

  { // stage 0: [16,320] x [320->64]
    v8f acc[4] = {};
    const _Float16* Wb = (const _Float16*)(ws + OFF_VF0);
#pragma unroll
    for (int kc = 0; kc < 10; ++kc) {
      v16h a = load_a_frag(stimh, KP, 0, kc * 32);
#pragma unroll
      for (int nt = 0; nt < 4; ++nt) acc[nt] = wmma16(a, load_b_frag(Wb, KP, nt * 16, kc * 32), acc[nt]);
    }
#pragma unroll
    for (int nt = 0; nt < 4; ++nt)
#pragma unroll
      for (int v = 0; v < 8; ++v) {
        int col = nt * 16 + c, r = v + hi8;
        buf0[r * 64 + col] = (_Float16)fmaxf(acc[nt][v] + bvf0[col], 0.0f);
      }
  }
  { // stage 1: [16,64] x [64->128]
    v8f acc[8] = {};
    const _Float16* Wb = (const _Float16*)(ws + OFF_VF1);
#pragma unroll
    for (int kc = 0; kc < 2; ++kc) {
      v16h a = load_a_frag(buf0, 64, 0, kc * 32);
#pragma unroll
      for (int nt = 0; nt < 8; ++nt) acc[nt] = wmma16(a, load_b_frag(Wb, 64, nt * 16, kc * 32), acc[nt]);
    }
#pragma unroll
    for (int nt = 0; nt < 8; ++nt)
#pragma unroll
      for (int v = 0; v < 8; ++v) {
        int col = nt * 16 + c, r = v + hi8;
        buf1[r * 128 + col] = (_Float16)fmaxf(acc[nt][v] + bvf1[col], 0.0f);
      }
  }
  { // stage 2: [16,128] x [128->64]
    v8f acc[4] = {};
    const _Float16* Wb = (const _Float16*)(ws + OFF_VF2);
#pragma unroll
    for (int kc = 0; kc < 4; ++kc) {
      v16h a = load_a_frag(buf1, 128, 0, kc * 32);
#pragma unroll
      for (int nt = 0; nt < 4; ++nt) acc[nt] = wmma16(a, load_b_frag(Wb, 128, nt * 16, kc * 32), acc[nt]);
    }
#pragma unroll
    for (int nt = 0; nt < 4; ++nt)
#pragma unroll
      for (int v = 0; v < 8; ++v) {
        int col = nt * 16 + c, r = v + hi8;
        buf0[r * 64 + col] = (_Float16)fmaxf(acc[nt][v] + bvf2[col], 0.0f);
      }
  }
  if (lane < 16) { // stage 3: 64 -> 1, tanh
    float sum = bvf3[0];
#pragma unroll 8
    for (int k = 0; k < 64; ++k) sum += (float)buf0[lane * 64 + k] * wvf3[k];
    vt_out[rowbase + lane] = fast_tanh(sum);
  }
}

// ---------------------------------------------------------------------------
// J/K GEMMs + 20-step FF recurrence + V_GPi_DP / V_D2 / lamd2 to workspace.
__global__ void __launch_bounds__(256) bg_ff(unsigned char* __restrict__ ws,
                     const float* __restrict__ bjd1, const float* __restrict__ bjd2,
                     const float* __restrict__ bkd1, const float* __restrict__ bkd2,
                     const float* __restrict__ deltavf,
                     const float* __restrict__ wd1gpi, const float* __restrict__ bd1gpi) {
  __shared__ float vd1s[8][16][20];
  const int lane = threadIdx.x & 31;
  const int wid  = threadIdx.x >> 5;
  const int rowbase = blockIdx.x * 128 + wid * 16;
  const _Float16* stimh = (const _Float16*)(ws + OFF_STIM) + (size_t)rowbase * KP;
  const _Float16* Wb = (const _Float16*)(ws + OFF_JKD);
  const int c = lane & 15, hi8 = (lane >> 4) * 8;
  __builtin_prefetch(Wb, 0, 0);

  v8f acc[8] = {};
#pragma unroll
  for (int kc = 0; kc < 10; ++kc) {
    v16h a = load_a_frag(stimh, KP, 0, kc * 32);
#pragma unroll
    for (int nt = 0; nt < 8; ++nt) acc[nt] = wmma16(a, load_b_frag(Wb, KP, nt * 16, kc * 32), acc[nt]);
  }

  float* vd2g = (float*)(ws + OFF_VD2);
  float* dpg  = (float*)(ws + OFF_DP);
  float* lamg = (float*)(ws + OFF_LAM);
  const bool ok1 = (16 + c) < 20;

#pragma unroll
  for (int v = 0; v < 8; ++v) {
    const int r = v + hi8;
    const int grow = rowbase + r;
    const float lam = sigmoidf_(deltavf[grow]);   // A1==A2==1, theta==0 -> lamd1==lamd2
    if (c == 0) lamg[grow] = lam;
    float J1[2], J2[2], K1[2], K2[2], v1[2] = {0.f, 0.f}, v2[2] = {0.f, 0.f};
    J1[0] = acc[0][v] + bjd1[c];  J2[0] = acc[2][v] + bjd2[c];
    K1[0] = acc[4][v] + bkd1[c];  K2[0] = acc[6][v] + bkd2[c];
    J1[1] = ok1 ? acc[1][v] + bjd1[16 + c] : 0.f;
    J2[1] = ok1 ? acc[3][v] + bjd2[16 + c] : 0.f;
    K1[1] = ok1 ? acc[5][v] + bkd1[16 + c] : 0.f;
    K2[1] = ok1 ? acc[7][v] + bkd2[16 + c] : 0.f;
    for (int s = 0; s < 20; ++s) {
#pragma unroll
      for (int q = 0; q < 2; ++q) {
        v1[q] = sigmoidf_(lam * (J1[q] * (1.f - v1[q]) + (1.f - K1[q]) * v1[q]));
        v2[q] = sigmoidf_(lam * (J2[q] * (1.f - v2[q]) + (1.f - K2[q]) * v2[q]));
      }
    }
    vd1s[wid][r][c] = v1[0];
    vd2g[(size_t)grow * 20 + c] = v2[0];
    if (ok1) { vd1s[wid][r][16 + c] = v1[1]; vd2g[(size_t)grow * 20 + 16 + c] = v2[1]; }
  }
  __syncthreads();
  { // V_GPi_DP = V_D1 @ w_d1gpi^T + b  (2 outputs per row)
    const int m = lane & 15, gi = lane >> 4;
    float sum = bd1gpi[gi];
#pragma unroll 4
    for (int j = 0; j < 20; ++j) sum += vd1s[wid][m][j] * wd1gpi[gi * 20 + j];
    dpg[(size_t)(rowbase + m) * 2 + gi] = sum;
  }
}

// ---------------------------------------------------------------------------
// 50-iteration STN/GPe/GPi + LSTM loop. One block owns 32 batch rows; w_slat
// and all recurrent state stay resident in LDS for the whole loop.
__global__ void __launch_bounds__(256) bg_stn(const unsigned char* __restrict__ ws,
                      const float* __restrict__ hx0, const float* __restrict__ cx0,
                      const float* __restrict__ bsg, const float* __restrict__ bglat,
                      const float* __restrict__ bgs, const float* __restrict__ bslat,
                      const float* __restrict__ bstngpi,
                      const float* __restrict__ bih, const float* __restrict__ bhh,
                      float* __restrict__ hx_out) {
  extern __shared__ unsigned char smem[];
  _Float16* s_slat  = (_Float16*)(smem + LDS_SLAT);
  _Float16* s_vstn  = (_Float16*)(smem + LDS_VSTN);
  float*    s_xstn  = (float*)(smem + LDS_XSTN);
  float*    s_gates = (float*)(smem + LDS_GATES);
  _Float16* s_xgpe  = (_Float16*)(smem + LDS_XGPE);
  _Float16* s_xh    = (_Float16*)(smem + LDS_XH);
  float*    s_c     = (float*)(smem + LDS_C);
  float*    s_h     = (float*)(smem + LDS_H);
  float*    s_vd2   = (float*)(smem + LDS_VD2);
  float*    s_vgpi  = (float*)(smem + LDS_VGPI);
  float*    s_dp    = (float*)(smem + LDS_DP);
  float*    s_lam   = (float*)(smem + LDS_LAM);

  const int tid = threadIdx.x, lane = tid & 31, wid = tid >> 5;
  const int mw = wid >> 2, g = wid & 3;
  const int m0 = mw * 16;
  const int rowbase = blockIdx.x * 32;
  const int c = lane & 15, hi8 = (lane >> 4) * 8;

  // --- stage w_slat into LDS with CDNA5 async DMA (ASYNCcnt-tracked) ---
  {
    const unsigned char* src = ws + OFF_SLAT;
    for (int i = tid; i < (NSLAT * KP * 2) / 16; i += 256) {
      unsigned lds_addr = (unsigned)(LDS_SLAT + i * 16);
      unsigned long long gaddr = (unsigned long long)(uintptr_t)(src + (size_t)i * 16);
      asm volatile("global_load_async_to_lds_b128 %0, %1, off"
                   :: "v"(lds_addr), "v"(gaddr) : "memory");
    }
    asm volatile("s_wait_asynccnt 0" ::: "memory");
  }
  // --- init recurrent state ---
  for (int i = tid; i < 32 * KP;    i += 256) s_vstn[i] = (_Float16)0.0f;
  for (int i = tid; i < 32 * NSLAT; i += 256) s_xstn[i] = 0.0f;
  for (int i = tid; i < 32 * 32;    i += 256) { s_xgpe[i] = (_Float16)0.0f; s_xh[i] = (_Float16)0.0f; }
  for (int i = tid; i < 32 * 2;     i += 256) s_vgpi[i] = 0.0f;
  { const float* dpg  = (const float*)(ws + OFF_DP);
    const float* lamg = (const float*)(ws + OFF_LAM);
    for (int i = tid; i < 32 * 2; i += 256) s_dp[i] = dpg[(size_t)rowbase * 2 + i];
    for (int i = tid; i < 32;     i += 256) s_lam[i] = lamg[rowbase + i]; }
  __syncthreads();
  { const float* vd2g = (const float*)(ws + OFF_VD2);
    for (int i = tid; i < 32 * 20; i += 256) {
      int r = i / 20, j = i - r * 20;
      float h0 = hx0[(size_t)(rowbase + r) * 20 + j];
      s_h[i] = h0;
      s_c[i] = cx0[(size_t)(rowbase + r) * 20 + j];
      s_xh[r * 32 + 2 + j] = (_Float16)h0;
      s_vd2[i] = vd2g[(size_t)(rowbase + r) * 20 + j];
    } }
  __syncthreads();

  const _Float16* Wsg = (const _Float16*)(ws + OFF_SG);
  const _Float16* Wgs = (const _Float16*)(ws + OFF_GS);
  const _Float16* Wgl = (const _Float16*)(ws + OFF_GLAT);
  const _Float16* Wsp = (const _Float16*)(ws + OFF_STNGPI);
  const _Float16* Wls = (const _Float16*)(ws + OFF_LSTM);

  for (int it = 0; it < 50; ++it) {
    // ---- phase 1: slat GEMM (all waves): out tiles nt = g, g+4, ... < 19 ----
    v8f acc[5] = {};
#pragma unroll
    for (int kc = 0; kc < 10; ++kc) {
      v16h a = load_a_frag(s_vstn, KP, m0, kc * 32);
#pragma unroll
      for (int t = 0; t < 5; ++t) {
        int nt = g + 4 * t;
        if (nt < 19) acc[t] = wmma16(a, load_b_frag(s_slat, KP, nt * 16, kc * 32), acc[t]);
      }
    }
    // ---- phase 1b: xgpe update on the spare slot (g==3 waves) ----
    if (g == 3) {
      v8f ag[2] = {};
#pragma unroll
      for (int kc = 0; kc < 10; ++kc) {
        v16h a = load_a_frag(s_vstn, KP, m0, kc * 32);
#pragma unroll
        for (int nt = 0; nt < 2; ++nt) ag[nt] = wmma16(a, load_b_frag(Wsg, KP, nt * 16, kc * 32), ag[nt]);
      }
      v16h a2 = load_a_frag(s_xgpe, 32, m0, 0);
#pragma unroll
      for (int nt = 0; nt < 2; ++nt) ag[nt] = wmma16(a2, load_b_frag(Wgl, 32, nt * 16, 0), ag[nt]);
      // ETA_GPE==1: xgpe_new = sg(vstn)+b_sg + glat(xgpe)+b_glat - V_D2
#pragma unroll
      for (int nt = 0; nt < 2; ++nt)
#pragma unroll
        for (int v = 0; v < 8; ++v) {
          int col = nt * 16 + c, rl = m0 + v + hi8;
          float val = 0.0f;
          if (col < 20) val = ag[nt][v] + bsg[col] + bglat[col] - s_vd2[rl * 20 + col];
          s_xgpe[rl * 32 + col] = (_Float16)val;
        }
    }
    __syncthreads();
    // ---- phase 2: gs GEMM folded into slat accumulators, then xstn/vstn update ----
    {
      v16h a = load_a_frag(s_xgpe, 32, m0, 0);
#pragma unroll
      for (int t = 0; t < 5; ++t) {
        int nt = g + 4 * t;
        if (nt < 19) acc[t] = wmma16(a, load_b_frag(Wgs, 32, nt * 16, 0), acc[t]);
      }
    }
#pragma unroll
    for (int t = 0; t < 5; ++t) {
      int nt = g + 4 * t;
      if (nt < 19) {
#pragma unroll
        for (int v = 0; v < 8; ++v) {
          int col = nt * 16 + c, rl = m0 + v + hi8;
          if (col < 300) {
            float sum = acc[t][v] + bgs[col] + bslat[col];
            float xo = s_xstn[rl * NSLAT + col];
            float xn = xo + (1.0f / 3.0f) * (-xo + sum);
            s_xstn[rl * NSLAT + col] = xn;
            s_vstn[rl * KP + col] = (_Float16)fast_tanh(s_lam[rl] * xn);
          }
        }
      }
    }
    __syncthreads();
    // ---- phase 3+4 (g==0 waves): stngpi GEMM -> vgpi -> LSTM gate GEMM.
    //      Same wave produces x=-vgpi and consumes it, so no barrier between. ----
    if (g == 0) {
      v8f ap = {};
#pragma unroll
      for (int kc = 0; kc < 10; ++kc) {
        v16h a = load_a_frag(s_vstn, KP, m0, kc * 32);
        ap = wmma16(a, load_b_frag(Wsp, KP, 0, kc * 32), ap);
      }
      if (c < 2) {
#pragma unroll
        for (int v = 0; v < 8; ++v) {
          int rl = m0 + v + hi8;
          float lam = s_lam[rl];
          float ip = lam * (ap[v] + bstngpi[c]);
          float vg = s_vgpi[rl * 2 + c];
          vg = vg + 0.1f * (-vg - s_dp[rl * 2 + c] + 2.0f * ip);
          s_vgpi[rl * 2 + c] = vg;
          s_xh[rl * 32 + c] = (_Float16)(-vg);   // LSTM input x = -vgpi
        }
      }
      // LSTM gate GEMM [16,(x|h) pad32] x [32->80]
      v8f al[5] = {};
      v16h a = load_a_frag(s_xh, 32, m0, 0);
#pragma unroll
      for (int t = 0; t < 5; ++t) al[t] = wmma16(a, load_b_frag(Wls, 32, t * 16, 0), al[t]);
#pragma unroll
      for (int t = 0; t < 5; ++t)
#pragma unroll
        for (int v = 0; v < 8; ++v) {
          int col = t * 16 + c, rl = m0 + v + hi8;
          s_gates[rl * 80 + col] = al[t][v] + bih[col] + bhh[col];
        }
    }
    __syncthreads();
    // ---- phase 5: LSTM elementwise (gate order i,f,g,o) ----
    for (int e = tid; e < 32 * 20; e += 256) {
      int r = e / 20, j = e - r * 20;
      float gi = s_gates[r * 80 + j];
      float gf = s_gates[r * 80 + 20 + j];
      float gg = s_gates[r * 80 + 40 + j];
      float go = s_gates[r * 80 + 60 + j];
      float cn = sigmoidf_(gf) * s_c[e] + sigmoidf_(gi) * fast_tanh(gg);
      float hn = sigmoidf_(go) * fast_tanh(cn);
      s_c[e] = cn; s_h[e] = hn;
      s_xh[r * 32 + 2 + j] = (_Float16)hn;
    }
    __syncthreads();
  }
  for (int e = tid; e < 32 * 20; e += 256) {
    int r = e / 20, j = e - r * 20;
    hx_out[(size_t)(rowbase + r) * 20 + j] = s_h[e];
  }
}

// ---------------------------------------------------------------------------
extern "C" void kernel_launch(void* const* d_in, const int* in_sizes, int n_in,
                              void* d_out, int out_size, void* d_ws, size_t ws_size,
                              hipStream_t stream) {
  const float* stimulus = (const float*)d_in[0];
  const float* deltavf  = (const float*)d_in[1];
  const float* hx0      = (const float*)d_in[2];
  const float* cx0      = (const float*)d_in[3];
  const float* w_vf0 = (const float*)d_in[4];   const float* b_vf0 = (const float*)d_in[5];
  const float* w_vf1 = (const float*)d_in[6];   const float* b_vf1 = (const float*)d_in[7];
  const float* w_vf2 = (const float*)d_in[8];   const float* b_vf2 = (const float*)d_in[9];
  const float* w_vf3 = (const float*)d_in[10];  const float* b_vf3 = (const float*)d_in[11];
  const float* w_jd1 = (const float*)d_in[12];  const float* b_jd1 = (const float*)d_in[13];
  const float* w_jd2 = (const float*)d_in[14];  const float* b_jd2 = (const float*)d_in[15];
  const float* w_kd1 = (const float*)d_in[16];  const float* b_kd1 = (const float*)d_in[17];
  const float* w_kd2 = (const float*)d_in[18];  const float* b_kd2 = (const float*)d_in[19];
  const float* w_sg  = (const float*)d_in[20];  const float* b_sg  = (const float*)d_in[21];
  const float* w_gs  = (const float*)d_in[22];  const float* b_gs  = (const float*)d_in[23];
  const float* w_glat= (const float*)d_in[24];  const float* b_glat= (const float*)d_in[25];
  const float* w_slat= (const float*)d_in[26];  const float* b_slat= (const float*)d_in[27];
  const float* w_d1gpi=(const float*)d_in[28];  const float* b_d1gpi=(const float*)d_in[29];
  const float* w_stngpi=(const float*)d_in[30]; const float* b_stngpi=(const float*)d_in[31];
  const float* w_ih  = (const float*)d_in[32];  const float* b_ih  = (const float*)d_in[33];
  const float* w_hh  = (const float*)d_in[34];  const float* b_hh  = (const float*)d_in[35];

  unsigned char* ws = (unsigned char*)d_ws;
  float* out = (float*)d_out;   // [B*20] hx, then [B] vt

  (void)hipFuncSetAttribute(reinterpret_cast<const void*>(bg_stn),
                            hipFuncAttributeMaxDynamicSharedMemorySize, STN_LDS);
  (void)hipFuncSetAttribute(reinterpret_cast<const void*>(bg_vf),
                            hipFuncAttributeMaxDynamicSharedMemorySize, 49152);

  bg_prep<<<512, 256, 0, stream>>>(stimulus, w_vf0, w_vf1, w_vf2,
                                   w_jd1, w_jd2, w_kd1, w_kd2,
                                   w_sg, w_gs, w_glat, w_slat,
                                   w_stngpi, w_ih, w_hh, ws);
  bg_vf<<<BATCH / 128, 256, 49152, stream>>>(ws, b_vf0, b_vf1, b_vf2, w_vf3, b_vf3,
                                             out + (size_t)BATCH * 20);
  bg_ff<<<BATCH / 128, 256, 0, stream>>>(ws, b_jd1, b_jd2, b_kd1, b_kd2,
                                         deltavf, w_d1gpi, b_d1gpi);
  bg_stn<<<BATCH / 32, 256, STN_LDS, stream>>>(ws, hx0, cx0, b_sg, b_glat, b_gs, b_slat,
                                               b_stngpi, b_ih, b_hh, out);
}
